// TRU_50294067036444
// MI455X (gfx1250) — matrix operand order repaired and checked by reading, not compiled, not run on hardware
//
#include <hip/hip_runtime.h>
#include <hip/hip_bf16.h>
#include <cstdint>

typedef __attribute__((ext_vector_type(16))) _Float16 v16h;
typedef __attribute__((ext_vector_type(8)))  _Float16 v8h;
typedef __attribute__((ext_vector_type(8)))  float    v8f;

union F16x16 { v16h v; v8h h8[2]; _Float16 e[16]; };

#define EPSV 1e-5f

__device__ __forceinline__ v8f zero8() {
  v8f z;
#pragma unroll
  for (int i = 0; i < 8; ++i) z[i] = 0.0f;
  return z;
}

__device__ __forceinline__ v8f wmma16(v16h a, v16h b, v8f c) {
  return __builtin_amdgcn_wmma_f32_16x16x32_f16(false, a, false, b, (short)0, c, false, false);
}

// A fragment for V_WMMA_F32_16X16X32_F16.
// A stored row-major [16 rows x K], row m contiguous in k.
// lane L<16 : row L,   halfs k0+0..15
// lane L>=16: row L-16, halfs k0+16..31
__device__ __forceinline__ v16h frag_a(const _Float16* base, int ld, int k0, int lane) {
  int r = lane & 15;
  const _Float16* p = base + r * ld + k0 + ((lane < 16) ? 0 : 16);
  F16x16 f;
  f.h8[0] = *(const v8h*)(p);
  f.h8[1] = *(const v8h*)(p + 8);
  return f.v;
}

// B fragment: B stored as Bt row-major [16 columns(n) x K], column n contiguous in k.
// lane L<16 : n=L,    halfs k0+0..15
// lane L>=16: n=L-16, halfs k0+16..31
__device__ __forceinline__ v16h frag_b(const _Float16* base, int ld, int k0, int lane) {
  int r = lane & 15;
  const _Float16* p = base + r * ld + k0 + ((lane < 16) ? 0 : 16);
  F16x16 f;
  f.h8[0] = *(const v8h*)(p);
  f.h8[1] = *(const v8h*)(p + 8);
  return f.v;
}

// ---------------------------------------------------------------------------
// Kernel 1: QKV 1x1 projections.
//   q -> qT[b][c][w][h], k -> kT[b][c][v][h], v -> vN[b][c][h][w]  (all f16)
// Grid: 8 batches * 256 pixel-tiles of 64; block 256 threads (8 waves).
// ---------------------------------------------------------------------------
__global__ __launch_bounds__(256) void qkv_kernel(
    const float* __restrict__ x1, const float* __restrict__ xc,
    const float* __restrict__ Wq, const float* __restrict__ bq,
    const float* __restrict__ Wk, const float* __restrict__ bk,
    const float* __restrict__ Wv, const float* __restrict__ bv,
    _Float16* __restrict__ qT, _Float16* __restrict__ kT,
    _Float16* __restrict__ vN) {
  __shared__ __align__(16) _Float16 sW[3 * 64 * 64];   // Wq|Wk|Wv as f16 [o][c]
  __shared__ __align__(16) _Float16 sXT[2][64][64];    // [pixel][c] for x1, xcur
  __shared__ float sB[3 * 64];

  const int tid = threadIdx.x;
  const int b = blockIdx.x >> 8;
  const int p0 = (blockIdx.x & 255) * 64;

  for (int i = tid; i < 4096; i += 256) {
    sW[i]        = (_Float16)Wq[i];
    sW[4096 + i] = (_Float16)Wk[i];
    sW[8192 + i] = (_Float16)Wv[i];
  }
  if (tid < 64) { sB[tid] = bq[tid]; sB[64 + tid] = bk[tid]; sB[128 + tid] = bv[tid]; }
  for (int i = tid; i < 4096; i += 256) {
    int c = i >> 6, j = i & 63;
    int g = ((b * 64 + c) << 14) + p0 + j;     // [b][c][pixel]
    sXT[0][j][c] = (_Float16)x1[g];
    sXT[1][j][c] = (_Float16)xc[g];
  }
  __syncthreads();

  const int wave = tid >> 5, lane = tid & 31;
  const int n0 = (wave & 3) * 16;    // pixel tile within the 64
  const int m0 = (wave >> 2) * 32;   // 2 out-channel tiles

  const int hi = (lane < 16) ? 0 : 8;
  const int n  = n0 + (lane & 15);
  const int pix = p0 + n;
  const int ph = pix >> 7, pw = pix & 127;

  for (int out = 0; out < 3; ++out) {
    const _Float16* Wb = sW + out * 4096;
    const _Float16* Xb = &sXT[out ? 1 : 0][0][0];
    v8f acc0 = zero8(), acc1 = zero8();
#pragma unroll
    for (int k0 = 0; k0 < 64; k0 += 32) {
      v16h bf = frag_b(Xb + n0 * 64, 64, k0, lane);
      v16h a0 = frag_a(Wb + m0 * 64, 64, k0, lane);
      v16h a1 = frag_a(Wb + (m0 + 16) * 64, 64, k0, lane);
      acc0 = wmma16(a0, bf, acc0);
      acc1 = wmma16(a1, bf, acc1);
    }
#pragma unroll
    for (int j = 0; j < 8; ++j) {
      int m = m0 + j + hi;
      float v0 = acc0[j] + sB[out * 64 + m];
      float v1 = acc1[j] + sB[out * 64 + m + 16];
      if (out == 0) {
        qT[((b * 64 + m) * 128 + pw) * 128 + ph]        = (_Float16)v0;
        qT[((b * 64 + m + 16) * 128 + pw) * 128 + ph]   = (_Float16)v1;
      } else if (out == 1) {
        kT[((b * 64 + m) * 128 + pw) * 128 + ph]        = (_Float16)v0;
        kT[((b * 64 + m + 16) * 128 + pw) * 128 + ph]   = (_Float16)v1;
      } else {
        vN[((b * 64 + m) << 14) + pix]                  = (_Float16)v0;
        vN[((b * 64 + m + 16) << 14) + pix]             = (_Float16)v1;
      }
    }
  }
}

// ---------------------------------------------------------------------------
// Kernel 2: fused attention per (b,c).
//   ET[v][w] = sum_h kT[v][h]*qT[w][h]; softmax rows (over w);
//   out[h][v] = sum_w V[h][w]*att[v][w]; y = gamma*out + x_cur -> yT[b][h][w][c]
// Grid: 512 blocks; block 256 threads; ~48KB LDS.
// ---------------------------------------------------------------------------
__global__ __launch_bounds__(256) void attn_kernel(
    const _Float16* __restrict__ qT, const _Float16* __restrict__ kT,
    const _Float16* __restrict__ vN, const float* __restrict__ xc,
    const float* __restrict__ gamma, _Float16* __restrict__ yT) {
  __shared__ __align__(16) _Float16 sA[128 * 32];     // kT / V k-chunk
  __shared__ __align__(16) _Float16 sBt[128 * 32];    // qT k-chunk
  __shared__ __align__(16) _Float16 sAtt[128 * 128];  // E^T then attention (f16)

  const int tid = threadIdx.x, lane = tid & 31, wave = tid >> 5;
  const int bc = blockIdx.x, b = bc >> 6, c = bc & 63;
  const _Float16* qTb = qT + (size_t)bc * 16384;
  const _Float16* kTb = kT + (size_t)bc * 16384;
  const _Float16* vBb = vN + (size_t)bc * 16384;

  const int mg = wave & 3;   // M tiles (2*mg+t)
  const int ng = wave >> 2;  // N tiles (4*ng+t)
  const int hi = (lane < 16) ? 0 : 8;
  const int nn = lane & 15;

  v8f acc[2][4];
#pragma unroll
  for (int i = 0; i < 2; ++i)
#pragma unroll
    for (int j = 0; j < 4; ++j) acc[i][j] = zero8();

  // ---- GEMM1: E^T = kT * qT^T, K streamed in chunks of 32 (over h) ----
  for (int ks = 0; ks < 4; ++ks) {
    int k0 = ks * 32;
    for (int i = tid; i < 512; i += 256) {
      int row = i >> 2, seg = (i & 3) * 8;
      *(v8h*)&sA[row * 32 + seg]  = *(const v8h*)&kTb[row * 128 + k0 + seg];
      *(v8h*)&sBt[row * 32 + seg] = *(const v8h*)&qTb[row * 128 + k0 + seg];
    }
    __syncthreads();
    v16h af[2], bf[4];
#pragma unroll
    for (int t = 0; t < 2; ++t) af[t] = frag_a(&sA[(2 * mg + t) * 512], 32, 0, lane);
#pragma unroll
    for (int t = 0; t < 4; ++t) bf[t] = frag_b(&sBt[(4 * ng + t) * 512], 32, 0, lane);
#pragma unroll
    for (int i = 0; i < 2; ++i)
#pragma unroll
      for (int j = 0; j < 4; ++j) acc[i][j] = wmma16(af[i], bf[j], acc[i][j]);
    __syncthreads();
  }
  // write E^T as f16 into sAtt
#pragma unroll
  for (int i = 0; i < 2; ++i)
#pragma unroll
    for (int j = 0; j < 4; ++j) {
      int w = (4 * ng + j) * 16 + nn;
#pragma unroll
      for (int e = 0; e < 8; ++e) {
        int v = (2 * mg + i) * 16 + e + hi;
        sAtt[v * 128 + w] = (_Float16)acc[i][j][e];
      }
    }
  __syncthreads();

  // ---- softmax over w for each row v (axis=2 of energy[b,c,w,v]) ----
  if (tid < 128) {
    float mx = -1e30f;
    for (int w = 0; w < 128; ++w) mx = fmaxf(mx, (float)sAtt[tid * 128 + w]);
    float sum = 0.0f;
    for (int w = 0; w < 128; ++w) {
      float e = __expf((float)sAtt[tid * 128 + w] - mx);
      sum += e;
      sAtt[tid * 128 + w] = (_Float16)e;
    }
    float inv = 1.0f / sum;
    for (int w = 0; w < 128; ++w)
      sAtt[tid * 128 + w] = (_Float16)((float)sAtt[tid * 128 + w] * inv);
  }
  __syncthreads();

  // ---- GEMM2: out[h][v] = V * att, K streamed over w ----
#pragma unroll
  for (int i = 0; i < 2; ++i)
#pragma unroll
    for (int j = 0; j < 4; ++j) acc[i][j] = zero8();
  for (int ks = 0; ks < 4; ++ks) {
    int k0 = ks * 32;
    for (int i = tid; i < 512; i += 256) {
      int row = i >> 2, seg = (i & 3) * 8;
      *(v8h*)&sA[row * 32 + seg] = *(const v8h*)&vBb[row * 128 + k0 + seg];
    }
    __syncthreads();
    v16h af[2], bf[4];
#pragma unroll
    for (int t = 0; t < 2; ++t) af[t] = frag_a(&sA[(2 * mg + t) * 512], 32, 0, lane);
#pragma unroll
    for (int t = 0; t < 4; ++t) bf[t] = frag_b(&sAtt[(4 * ng + t) * 16 * 128], 128, k0, lane);
#pragma unroll
    for (int i = 0; i < 2; ++i)
#pragma unroll
      for (int j = 0; j < 4; ++j) acc[i][j] = wmma16(af[i], bf[j], acc[i][j]);
    __syncthreads();
  }

  // ---- epilogue: y = gamma*out + x_cur -> yT[b][h][w][c] (f16) ----
  const float g = gamma[0];
#pragma unroll
  for (int i = 0; i < 2; ++i)
#pragma unroll
    for (int j = 0; j < 4; ++j) {
      int w = (4 * ng + j) * 16 + nn;
#pragma unroll
      for (int e = 0; e < 8; ++e) {
        int h = (2 * mg + i) * 16 + e + hi;
        float y = g * acc[i][j][e] + xc[((b * 64 + c) << 14) + h * 128 + w];
        yT[((b * 128 + h) * 128 + w) * 64 + c] = (_Float16)y;
      }
    }
}

// ---------------------------------------------------------------------------
// Kernel 3: 3x3 conv (SAME) as 9 shifted 1x1 GEMMs + BN1 + ReLU.
//   yT[b][h][w][c] -> z[b][h][w][o]  (f16)
// Grid: B*H = 1024 blocks; each wave: 16 pixels x 64 out channels.
// ---------------------------------------------------------------------------
__global__ __launch_bounds__(256) void conv1_kernel(
    const _Float16* __restrict__ yT, const float* __restrict__ w1,
    const float* __restrict__ sc, const float* __restrict__ bi,
    const float* __restrict__ mn, const float* __restrict__ vr,
    _Float16* __restrict__ z) {
  __shared__ __align__(16) _Float16 sWs[64 * 64];  // weight slice [o][c] for shift
  __shared__ float sInv[64], sBeta[64];

  const int tid = threadIdx.x, lane = tid & 31, wave = tid >> 5;
  const int b = blockIdx.x >> 7, h = blockIdx.x & 127;
  const int wsp = wave * 16 + (lane & 15);

  if (tid < 64) {
    float inv = sc[tid] * rsqrtf(vr[tid] + EPSV);
    sInv[tid] = inv;
    sBeta[tid] = bi[tid] - mn[tid] * inv;
  }

  v8f acc[4];
#pragma unroll
  for (int mt = 0; mt < 4; ++mt) acc[mt] = zero8();

  for (int s = 0; s < 9; ++s) {
    const int dy = s / 3 - 1, dx = s % 3 - 1;
    __syncthreads();
    for (int i = tid; i < 4096; i += 256)   // i = o*64 + c
      sWs[i] = (_Float16)w1[i * 9 + (dy + 1) * 3 + (dx + 1)];
    __syncthreads();

    const int hh = h + dy;                 // uniform across block
    if (hh < 0 || hh >= 128) continue;
    const int ww = wsp + dx;
    const bool okw = ((unsigned)ww < 128u);
    const _Float16* pb = yT + (((size_t)(b * 128 + hh) * 128 + ww) * 64);
#pragma unroll
    for (int k0 = 0; k0 < 64; k0 += 32) {
      F16x16 f;
      if (okw) {
        const _Float16* p = pb + k0 + ((lane < 16) ? 0 : 16);
        f.h8[0] = *(const v8h*)p;
        f.h8[1] = *(const v8h*)(p + 8);
      } else {
#pragma unroll
        for (int e = 0; e < 16; ++e) f.e[e] = (_Float16)0.0f;
      }
#pragma unroll
      for (int mt = 0; mt < 4; ++mt) {
        v16h a = frag_a(&sWs[mt * 16 * 64], 64, k0, lane);
        acc[mt] = wmma16(a, f.v, acc[mt]);
      }
    }
  }

  const int hi = (lane < 16) ? 0 : 8;
  _Float16* zp = z + (((size_t)(b * 128 + h) * 128 + wsp) * 64);
#pragma unroll
  for (int mt = 0; mt < 4; ++mt)
#pragma unroll
    for (int e = 0; e < 8; ++e) {
      int o = mt * 16 + e + hi;
      float val = acc[mt][e] * sInv[o] + sBeta[o];
      zp[o] = (_Float16)fmaxf(val, 0.0f);
    }
}

// ---------------------------------------------------------------------------
// Kernel 4: 1x1 conv + BN2 + ReLU -> fp32 NCHW output.
// ---------------------------------------------------------------------------
__global__ __launch_bounds__(256) void conv2_kernel(
    const _Float16* __restrict__ z, const float* __restrict__ w2,
    const float* __restrict__ sc, const float* __restrict__ bi,
    const float* __restrict__ mn, const float* __restrict__ vr,
    float* __restrict__ outp) {
  __shared__ __align__(16) _Float16 sWs[64 * 64];
  __shared__ float sInv[64], sBeta[64];

  const int tid = threadIdx.x, lane = tid & 31, wave = tid >> 5;
  const int b = blockIdx.x >> 7, h = blockIdx.x & 127;
  const int wsp = wave * 16 + (lane & 15);

  for (int i = tid; i < 4096; i += 256) sWs[i] = (_Float16)w2[i];
  if (tid < 64) {
    float inv = sc[tid] * rsqrtf(vr[tid] + EPSV);
    sInv[tid] = inv;
    sBeta[tid] = bi[tid] - mn[tid] * inv;
  }
  __syncthreads();

  v8f acc[4];
#pragma unroll
  for (int mt = 0; mt < 4; ++mt) acc[mt] = zero8();

  const _Float16* pb = z + (((size_t)(b * 128 + h) * 128 + wsp) * 64);
#pragma unroll
  for (int k0 = 0; k0 < 64; k0 += 32) {
    F16x16 f;
    const _Float16* p = pb + k0 + ((lane < 16) ? 0 : 16);
    f.h8[0] = *(const v8h*)p;
    f.h8[1] = *(const v8h*)(p + 8);
#pragma unroll
    for (int mt = 0; mt < 4; ++mt) {
      v16h a = frag_a(&sWs[mt * 16 * 64], 64, k0, lane);
      acc[mt] = wmma16(a, f.v, acc[mt]);
    }
  }

  const int hi = (lane < 16) ? 0 : 8;
#pragma unroll
  for (int mt = 0; mt < 4; ++mt)
#pragma unroll
    for (int e = 0; e < 8; ++e) {
      int o = mt * 16 + e + hi;
      float val = acc[mt][e] * sInv[o] + sBeta[o];
      outp[((size_t)(b * 64 + o) * 128 + h) * 128 + wsp] = fmaxf(val, 0.0f);
    }
}

// ---------------------------------------------------------------------------
extern "C" void kernel_launch(void* const* d_in, const int* in_sizes, int n_in,
                              void* d_out, int out_size, void* d_ws, size_t ws_size,
                              hipStream_t stream) {
  const float* x1   = (const float*)d_in[0];
  const float* xc   = (const float*)d_in[1];
  const float* Wq   = (const float*)d_in[2];
  const float* bq   = (const float*)d_in[3];
  const float* Wk   = (const float*)d_in[4];
  const float* bk   = (const float*)d_in[5];
  const float* Wv   = (const float*)d_in[6];
  const float* bv   = (const float*)d_in[7];
  const float* gma  = (const float*)d_in[8];
  const float* w1   = (const float*)d_in[9];
  const float* bn1s = (const float*)d_in[10];
  const float* bn1b = (const float*)d_in[11];
  const float* bn1m = (const float*)d_in[12];
  const float* bn1v = (const float*)d_in[13];
  const float* w2   = (const float*)d_in[14];
  const float* bn2s = (const float*)d_in[15];
  const float* bn2b = (const float*)d_in[16];
  const float* bn2m = (const float*)d_in[17];
  const float* bn2v = (const float*)d_in[18];

  const size_t SLAB = (size_t)8 * 64 * 128 * 128;  // 8,388,608 f16 elements
  _Float16* qT = (_Float16*)d_ws;
  _Float16* kT = qT + SLAB;
  _Float16* vN = kT + SLAB;
  _Float16* yT = vN + SLAB;
  _Float16* z  = qT;  // qT slab is dead after attn_kernel; reuse for z

  qkv_kernel<<<2048, 256, 0, stream>>>(x1, xc, Wq, bq, Wk, bk, Wv, bv, qT, kT, vN);
  attn_kernel<<<512, 256, 0, stream>>>(qT, kT, vN, xc, gma, yT);
  conv1_kernel<<<1024, 256, 0, stream>>>(yT, w1, bn1s, bn1b, bn1m, bn1v, z);
  conv2_kernel<<<1024, 256, 0, stream>>>(z, w2, bn2s, bn2b, bn2m, bn2v, (float*)d_out);
}
